// VSSBlock_18150531793699
// MI455X (gfx1250) — matrix operand order repaired and checked by reading, not compiled, not run on hardware
//
#include <hip/hip_runtime.h>
#include <math.h>
#include <stdint.h>

// ---------------------------------------------------------------------------
// VSS block (VMamba-style) for MI455X / gfx1250, wave32 + WMMA f16->f32.
// GEMMs: v_wmma_f32_16x16x32_f16, double-buffered LDS tiles fed by
// GLOBAL_LOAD_ASYNC_TO_LDS_B128 (ASYNCcnt) when the toolchain exposes it.
// ---------------------------------------------------------------------------

typedef __attribute__((ext_vector_type(16))) _Float16 v16h_t;
typedef __attribute__((ext_vector_type(8)))  _Float16 v8h_t;
typedef __attribute__((ext_vector_type(8)))  float    v8f_t;
typedef int v4i_gcc __attribute__((vector_size(16)));   // int4, GCC-vector style

#if defined(__gfx1250__) && __has_builtin(__builtin_amdgcn_global_load_async_to_lds_b128)
#define HAS_ASYNC_LDS 1
#else
#define HAS_ASYNC_LDS 0
#endif

// Problem constants (from reference)
constexpr int Bc  = 8;
constexpr int Hc  = 56;
constexpr int Wc  = 56;
constexpr int Dc  = 192;
constexpr int Nc  = Hc * Wc;       // 3136
constexpr int DIc = 384;           // D * EXP
constexpr int MHc = 768;           // D * 4
constexpr int TOK  = Bc * Nc;      // 25088 tokens
constexpr int STOK = 4 * Bc * Nc;  // 100352 scan tokens

__device__ __forceinline__ float waveSum(float v) {
#pragma unroll
  for (int o = 16; o > 0; o >>= 1) v += __shfl_xor(v, o, 32);
  return v;
}

// Block-wide sum for NW waves (NW <= 32). `red` must hold NW floats.
template <int NW>
__device__ __forceinline__ float blockSum(float v, float* red) {
  const int wv = threadIdx.x >> 5, ln = threadIdx.x & 31;
  v = waveSum(v);
  __syncthreads();                 // protect red from previous use
  if (ln == 0) red[wv] = v;
  __syncthreads();
  float t = (ln < NW) ? red[ln] : 0.f;
  return waveSum(t);
}

__device__ __forceinline__ float siluf(float x) { return x / (1.f + expf(-x)); }
__device__ __forceinline__ float geluf(float x) {
  return 0.5f * x * (1.f + erff(x * 0.70710678118654752f));
}

// 16-byte global -> LDS copy. Async path bypasses VGPRs (ASYNCcnt tracked).
__device__ __forceinline__ void copy16_g2l(const _Float16* g, _Float16* l) {
#if HAS_ASYNC_LDS
  __builtin_amdgcn_global_load_async_to_lds_b128(
      (__attribute__((address_space(1))) v4i_gcc*)(uintptr_t)g,
      (__attribute__((address_space(3))) v4i_gcc*)(uint32_t)(uintptr_t)l,
      /*imm offset*/ 0, /*cpol*/ 0);
#else
  *(v8h_t*)l = *(const v8h_t*)g;
#endif
}

__device__ __forceinline__ void wait_async_copies() {
#if HAS_ASYNC_LDS
#if __has_builtin(__builtin_amdgcn_s_wait_asynccnt)
  __builtin_amdgcn_s_wait_asynccnt(0);
#else
  asm volatile("s_wait_asynccnt 0x0" ::: "memory");
#endif
#endif
}

// ---------------------------------------------------------------------------
// Weight convert + transpose: dst[n*K + k] = (f16)src[k*N + n]
// ---------------------------------------------------------------------------
__global__ __launch_bounds__(256) void transpose_f16(const float* __restrict__ src,
                                                     _Float16* __restrict__ dst,
                                                     int K, int N) {
  int idx = blockIdx.x * 256 + threadIdx.x;
  if (idx >= K * N) return;
  int k = idx / N, n = idx % N;
  dst[(size_t)n * K + k] = (_Float16)src[idx];
}

// ---------------------------------------------------------------------------
// LN1 over D=192, one wave per token (lane handles 6 channels).
// ---------------------------------------------------------------------------
__global__ __launch_bounds__(256) void ln1_kernel(const float* __restrict__ x,
                                                  const float* __restrict__ g,
                                                  const float* __restrict__ b,
                                                  float* __restrict__ out) {
  const int token = blockIdx.x * 8 + (threadIdx.x >> 5);
  const int lane  = threadIdx.x & 31;
  const float* row = x + (size_t)token * Dc;
  float v[6], s = 0.f;
#pragma unroll
  for (int i = 0; i < 6; ++i) { v[i] = row[lane + i * 32]; s += v[i]; }
  const float mean = waveSum(s) * (1.f / Dc);
  float var = 0.f;
#pragma unroll
  for (int i = 0; i < 6; ++i) { float d = v[i] - mean; var += d * d; }
  var = waveSum(var) * (1.f / Dc);
  const float r = rsqrtf(var + 1e-5f);
  float* orow = out + (size_t)token * Dc;
#pragma unroll
  for (int i = 0; i < 6; ++i) {
    int c = lane + i * 32;
    orow[c] = (v[i] - mean) * r * g[c] + b[c];
  }
}

// ---------------------------------------------------------------------------
// 3x3 depthwise conv (pad 1) + scatter into 4 scan orders (f16).
// scans layout: [(g*B + b) * N + n] * D + d  (g = scan id 0..3)
// ---------------------------------------------------------------------------
__global__ __launch_bounds__(256) void dwconv_scan(const float* __restrict__ xl,
                                                   const float* __restrict__ dww,
                                                   const float* __restrict__ dwb,
                                                   _Float16* __restrict__ scans) {
  size_t e = (size_t)blockIdx.x * 256 + threadIdx.x;   // < B*H*W*D
  const int d = (int)(e % Dc);
  size_t r = e / Dc;
  const int w = (int)(r % Wc); r /= Wc;
  const int h = (int)(r % Hc);
  const int b = (int)(r / Hc);

  float acc = dwb[d];
#pragma unroll
  for (int ky = 0; ky < 3; ++ky) {
    const int hh = h + ky - 1;
    if (hh < 0 || hh >= Hc) continue;
#pragma unroll
    for (int kx = 0; kx < 3; ++kx) {
      const int ww = w + kx - 1;
      if (ww < 0 || ww >= Wc) continue;
      acc += xl[(((size_t)b * Hc + hh) * Wc + ww) * Dc + d] * dww[d * 9 + ky * 3 + kx];
    }
  }
  const _Float16 hv = (_Float16)acc;
  const int n0 = h * Wc + w;
  const int n1 = (Hc - 1 - h) * Wc + (Wc - 1 - w);
  const int n2 = w * Hc + h;                 // transpose scan (H==W)
  const int n3 = (Wc - 1 - w) * Hc + (Hc - 1 - h);
  scans[(((size_t)(0 * Bc + b)) * Nc + n0) * Dc + d] = hv;
  scans[(((size_t)(1 * Bc + b)) * Nc + n1) * Dc + d] = hv;
  scans[(((size_t)(2 * Bc + b)) * Nc + n2) * Dc + d] = hv;
  scans[(((size_t)(3 * Bc + b)) * Nc + n3) * Dc + d] = hv;
}

// ---------------------------------------------------------------------------
// WMMA GEMM: C[M,N] = A[M,K] * BT[N,K]^T, f16 in, f32 accumulate.
// Block: 256 threads (8 waves in a 2x4 wave grid). Tile BM x BN, BK = 32.
// Double-buffered LDS; async global->LDS copies overlap next-tile fetch with
// WMMA on the current tile.
// EPI: 0 = store f16, 1 = store f32, 2 = (acc+bias) GELU -> f16,
//      3 = acc + bias + resid -> f32
// ---------------------------------------------------------------------------
template <int BM, int BN, int EPI>
__global__ __launch_bounds__(256) void gemm_wmma(const _Float16* __restrict__ A,
                                                 const _Float16* __restrict__ BT,
                                                 int M, int N, int K,
                                                 float* __restrict__ outF,
                                                 _Float16* __restrict__ outH,
                                                 const float* __restrict__ bias,
                                                 const float* __restrict__ resid) {
  constexpr int BK = 32;
  constexpr int FM = BM / 32;   // 16x16 frags per wave, M dir (2 wave rows)
  constexpr int FN = BN / 64;   // 16x16 frags per wave, N dir (4 wave cols)
  constexpr int ACH = (BM * BK) / (256 * 8);   // 8-half chunks per thread (A)
  constexpr int BCH = (BN * BK) / (256 * 8);   // 8-half chunks per thread (B)

  __shared__ alignas(32) _Float16 Alds[2][BM * BK];
  __shared__ alignas(32) _Float16 Blds[2][BN * BK];

  const int tid  = threadIdx.x;
  const int lane = tid & 31;
  const int wave = tid >> 5;
  const int wr   = wave >> 2;       // 0..1
  const int wc   = wave & 3;        // 0..3
  const int half = lane >> 4;       // 0..1 (half-wave)
  const int l16  = lane & 15;

  const int m0 = blockIdx.x * BM;
  const int n0 = blockIdx.y * BN;

  v8f_t acc[FM][FN];
#pragma unroll
  for (int i = 0; i < FM; ++i)
#pragma unroll
    for (int j = 0; j < FN; ++j) acc[i][j] = (v8f_t)0.f;

  // Stage one BK-slice of A and BT into LDS buffer `buf` (async when possible)
  auto stage = [&](int buf, int k0) {
#pragma unroll
    for (int c = 0; c < ACH; ++c) {
      const int cid = tid + c * 256;
      const int row = cid >> 2;            // BK/8 = 4 chunks per row
      const int kc  = (cid & 3) * 8;
      copy16_g2l(&A[(size_t)(m0 + row) * K + k0 + kc], &Alds[buf][row * BK + kc]);
    }
#pragma unroll
    for (int c = 0; c < BCH; ++c) {
      const int cid = tid + c * 256;
      const int row = cid >> 2;
      const int kc  = (cid & 3) * 8;
      copy16_g2l(&BT[(size_t)(n0 + row) * K + k0 + kc], &Blds[buf][row * BK + kc]);
    }
  };

  int buf = 0;
  stage(buf, 0);
  wait_async_copies();
  __syncthreads();

  for (int k0 = 0; k0 < K; k0 += BK) {
    // Kick off next tile into the other buffer while we compute this one.
    if (k0 + BK < K) {
      stage(buf ^ 1, k0 + BK);
      // L2 prefetch two tiles ahead (global_prefetch_b8)
      if (k0 + 2 * BK < K) {
        __builtin_prefetch(&A[(size_t)(m0 + (tid >> 2)) * K + (k0 + 2 * BK) + (tid & 3) * 8], 0, 1);
        __builtin_prefetch(&BT[(size_t)(n0 + ((tid >> 2) % BN)) * K + (k0 + 2 * BK) + (tid & 3) * 8], 0, 1);
      }
    }

    // B fragments: lane holds column N = l16, K = e + 16*half  (contiguous run)
    v16h_t bfrag[FN];
#pragma unroll
    for (int fn = 0; fn < FN; ++fn) {
      const int n = wc * (16 * FN) + fn * 16 + l16;
      bfrag[fn] = *(const v16h_t*)&Blds[buf][n * BK + half * 16];
    }
    // A fragments: lane holds row M = l16, K = e + 16*half  (contiguous run)
#pragma unroll
    for (int fm = 0; fm < FM; ++fm) {
      const int mr = wr * (16 * FM) + fm * 16 + l16;
      const v8h_t lo = *(const v8h_t*)&Alds[buf][mr * BK + half * 16];
      const v8h_t hi = *(const v8h_t*)&Alds[buf][mr * BK + half * 16 + 8];
      const v16h_t afrag = __builtin_shufflevector(
          lo, hi, 0, 1, 2, 3, 4, 5, 6, 7, 8, 9, 10, 11, 12, 13, 14, 15);
#pragma unroll
      for (int fn = 0; fn < FN; ++fn) {
        acc[fm][fn] = __builtin_amdgcn_wmma_f32_16x16x32_f16(
            false, afrag, false, bfrag[fn], (short)0, acc[fm][fn], false, false);
      }
    }

    // Wait for this wave's async copies into buf^1, then block-wide barrier.
    wait_async_copies();
    __syncthreads();
    buf ^= 1;
  }

  // Epilogue. C/D layout: VGPR r -> (M = r + 8*half, N = l16)
#pragma unroll
  for (int fm = 0; fm < FM; ++fm)
#pragma unroll
    for (int fn = 0; fn < FN; ++fn)
#pragma unroll
      for (int r = 0; r < 8; ++r) {
        const int row = m0 + wr * (16 * FM) + fm * 16 + half * 8 + r;
        const int col = n0 + wc * (16 * FN) + fn * 16 + l16;
        const size_t o = (size_t)row * N + col;
        const float v = acc[fm][fn][r];
        if constexpr (EPI == 0) {
          outH[o] = (_Float16)v;
        } else if constexpr (EPI == 1) {
          outF[o] = v;
        } else if constexpr (EPI == 2) {
          outH[o] = (_Float16)geluf(v + bias[col]);
        } else {
          outF[o] = v + bias[col] + resid[o];
        }
      }
}

// ---------------------------------------------------------------------------
// Mamba middle: causal depthwise conv1d (K=4) + SiLU + *mD + LN(384) + gate.
// One block per scan token, 384 threads (12 waves).
// xz rows: [x_(0..383) | z(384..767)]
// ---------------------------------------------------------------------------
__global__ __launch_bounds__(384) void mamba_mid(const _Float16* __restrict__ xz,
                                                 const float* __restrict__ convw,
                                                 const float* __restrict__ convb,
                                                 const float* __restrict__ mDp,
                                                 const float* __restrict__ mng,
                                                 const float* __restrict__ mnb,
                                                 _Float16* __restrict__ a2) {
  __shared__ float red[12];
  const int t = blockIdx.x;          // scan token (< STOK)
  const int d = threadIdx.x;         // channel (< 384)
  const int seq = t / Nc;            // which (scan, batch) sequence
  const int n   = t % Nc;            // position in sequence

  float xc = convb[d];
#pragma unroll
  for (int k = 0; k < 4; ++k) {
    const int nn = n - 3 + k;
    if (nn >= 0)
      xc += (float)xz[((size_t)seq * Nc + nn) * (2 * DIc) + d] * convw[d * 4 + k];
  }
  const float u = siluf(xc) * mDp[d];

  const float mean = blockSum<12>(u, red) * (1.f / DIc);
  const float dv = u - mean;
  const float var = blockSum<12>(dv * dv, red) * (1.f / DIc);
  const float ln = dv * rsqrtf(var + 1e-5f) * mng[d] + mnb[d];

  const float z = (float)xz[(size_t)t * (2 * DIc) + DIc + d];
  a2[(size_t)t * DIc + d] = (_Float16)(ln * siluf(z));
}

// ---------------------------------------------------------------------------
// Combine 4 scan outputs + residual; write x_new (f32) and LN2(x_new) (f16).
// One block per token, 192 threads (6 waves).
// ---------------------------------------------------------------------------
__global__ __launch_bounds__(192) void combine_ln2(const float* __restrict__ ys,
                                                   const float* __restrict__ x,
                                                   const float* __restrict__ g2,
                                                   const float* __restrict__ b2,
                                                   float* __restrict__ xnew,
                                                   _Float16* __restrict__ a3) {
  __shared__ float red[6];
  const int t = blockIdx.x;         // token (< TOK)
  const int d = threadIdx.x;        // channel (< 192)
  const int b = t / Nc;
  const int n = t % Nc;
  const int h = n / Wc, w = n % Wc;

  const float y0 = ys[(((size_t)(0 * Bc + b)) * Nc + (h * Wc + w)) * Dc + d];
  const float y1 = ys[(((size_t)(1 * Bc + b)) * Nc + ((Hc - 1 - h) * Wc + (Wc - 1 - w))) * Dc + d];
  const float y2 = ys[(((size_t)(2 * Bc + b)) * Nc + (w * Wc + h)) * Dc + d];
  const float y3 = ys[(((size_t)(3 * Bc + b)) * Nc + ((Hc - 1 - w) * Wc + (Wc - 1 - h))) * Dc + d];

  const float xn = x[(size_t)t * Dc + d] + 0.25f * (y0 + y1 + y2 + y3);
  xnew[(size_t)t * Dc + d] = xn;

  const float mean = blockSum<6>(xn, red) * (1.f / Dc);
  const float dv = xn - mean;
  const float var = blockSum<6>(dv * dv, red) * (1.f / Dc);
  a3[(size_t)t * Dc + d] = (_Float16)(dv * rsqrtf(var + 1e-5f) * g2[d] + b2[d]);
}

// ---------------------------------------------------------------------------
extern "C" void kernel_launch(void* const* d_in, const int* in_sizes, int n_in,
                              void* d_out, int out_size, void* d_ws, size_t ws_size,
                              hipStream_t stream) {
  const float* x      = (const float*)d_in[0];
  // d_in[1]=H, d_in[2]=W (compile-time constants here)
  const float* ln1_g  = (const float*)d_in[3];
  const float* ln1_b  = (const float*)d_in[4];
  const float* dw_w   = (const float*)d_in[5];
  const float* dw_b   = (const float*)d_in[6];
  const float* in_w   = (const float*)d_in[7];
  const float* conv_w = (const float*)d_in[8];
  const float* conv_b = (const float*)d_in[9];
  const float* mD     = (const float*)d_in[10];
  const float* mn_g   = (const float*)d_in[11];
  const float* mn_b   = (const float*)d_in[12];
  const float* out_w  = (const float*)d_in[13];
  const float* ln2_g  = (const float*)d_in[14];
  const float* ln2_b  = (const float*)d_in[15];
  const float* w1     = (const float*)d_in[16];
  const float* b1     = (const float*)d_in[17];
  const float* w2     = (const float*)d_in[18];
  const float* b2     = (const float*)d_in[19];
  float* out = (float*)d_out;

  char* base = (char*)d_ws;
  size_t off = 0;
  auto alloc = [&](size_t bytes) -> void* {
    void* p = base + off;
    off = (off + bytes + 255) & ~(size_t)255;
    return p;
  };
  _Float16* wT1  = (_Float16*)alloc((size_t)MHc * Dc * 2);     // in_w^T  (768x192)
  _Float16* wT2  = (_Float16*)alloc((size_t)Dc * DIc * 2);     // out_w^T (192x384)
  _Float16* wT3  = (_Float16*)alloc((size_t)MHc * Dc * 2);     // w1^T    (768x192)
  _Float16* wT4  = (_Float16*)alloc((size_t)Dc * MHc * 2);     // w2^T    (192x768)
  float*    xl   = (float*)alloc((size_t)TOK * Dc * 4);        // LN1(x)
  _Float16* scans= (_Float16*)alloc((size_t)STOK * Dc * 2);    // 4 scan orders
  _Float16* xz   = (_Float16*)alloc((size_t)STOK * (2 * DIc) * 2);
  _Float16* a2   = (_Float16*)alloc((size_t)STOK * DIc * 2);   // gated ssm feats
  float*    ys   = (float*)alloc((size_t)STOK * Dc * 4);       // per-scan outputs
  float*    xnew = (float*)alloc((size_t)TOK * Dc * 4);        // residual stream
  _Float16* a3   = (_Float16*)alloc((size_t)TOK * Dc * 2);     // LN2(x_new)
  _Float16* hg   = (_Float16*)alloc((size_t)TOK * MHc * 2);    // gelu(h@w1+b1)

  // Weight convert + transpose (N x K, f16)
  transpose_f16<<<(Dc * MHc + 255) / 256, 256, 0, stream>>>(in_w, wT1, Dc, MHc);
  transpose_f16<<<(DIc * Dc + 255) / 256, 256, 0, stream>>>(out_w, wT2, DIc, Dc);
  transpose_f16<<<(Dc * MHc + 255) / 256, 256, 0, stream>>>(w1, wT3, Dc, MHc);
  transpose_f16<<<(MHc * Dc + 255) / 256, 256, 0, stream>>>(w2, wT4, MHc, Dc);

  // LN1
  ln1_kernel<<<TOK / 8, 256, 0, stream>>>(x, ln1_g, ln1_b, xl);
  // dw 3x3 conv + 4-way scan scatter
  dwconv_scan<<<(TOK * Dc) / 256, 256, 0, stream>>>(xl, dw_w, dw_b, scans);
  // GEMM1: xz = scans @ in_w   (100352 x 768, K=192)
  gemm_wmma<128, 128, 0><<<dim3(STOK / 128, (2 * DIc) / 128), 256, 0, stream>>>(
      scans, wT1, STOK, 2 * DIc, Dc, nullptr, xz, nullptr, nullptr);
  // conv1d + silu + LN + gate
  mamba_mid<<<STOK, 384, 0, stream>>>(xz, conv_w, conv_b, mD, mn_g, mn_b, a2);
  // GEMM2: ys = a2 @ out_w     (100352 x 192, K=384)
  gemm_wmma<128, 64, 1><<<dim3(STOK / 128, Dc / 64), 256, 0, stream>>>(
      a2, wT2, STOK, Dc, DIc, ys, nullptr, nullptr, nullptr);
  // merge scans + residual + LN2
  combine_ln2<<<TOK, 192, 0, stream>>>(ys, x, ln2_g, ln2_b, xnew, a3);
  // MLP GEMM1: hg = gelu(a3 @ w1 + b1)   (25088 x 768, K=192)
  gemm_wmma<128, 128, 2><<<dim3(TOK / 128, MHc / 128), 256, 0, stream>>>(
      a3, wT3, TOK, MHc, Dc, nullptr, hg, b1, nullptr);
  // MLP GEMM2: out = xnew + hg @ w2 + b2  (25088 x 192, K=768)
  gemm_wmma<128, 64, 3><<<dim3(TOK / 128, Dc / 64), 256, 0, stream>>>(
      hg, wT4, TOK, Dc, MHc, out, nullptr, b2, xnew);
}
